// Main_88862873354439
// MI455X (gfx1250) — matrix operand order repaired and checked by reading, not compiled, not run on hardware
//
#include <hip/hip_runtime.h>
#include <hip/hip_bf16.h>

// ---------------------------------------------------------------------------
// 2-layer LSTM scan (SEQ=65536, HID=128, VOCAB=128) + FC, for MI455X / gfx1250.
//
// K1: GA = x @ WA1^T + bA1 + bA2   (parallel GEMM, v_wmma_f32_16x16x32_f16)
// K2: single-WGP sequential scan: 16 WMMAs / timestep critical path,
//     weights resident in LDS (f16, padded pitch -> bank-conflict-free
//     ds_load_b128) and in registers (WB2 as pre-swizzled B operands).
// Workspace use: GA as f16 = 65536*512*2 = 64 MB.
// ---------------------------------------------------------------------------

typedef __attribute__((ext_vector_type(16))) _Float16 v16h;
typedef __attribute__((ext_vector_type(8)))  _Float16 v8h;
typedef __attribute__((ext_vector_type(8)))  float    v8f;

#define SEQ    65536
#define HID    128
#define VOCAB  128
#define GATES  512
#define WPITCH 136   // f16 elems per LDS weight row: 272B pitch, 16B aligned,
                     // dword index 68*n -> bank 4n mod 64 (conflict-free b128)

__device__ __forceinline__ float sigm(float x)  { return 1.0f / (1.0f + __expf(-x)); }
__device__ __forceinline__ float tanhfast(float x){ return 2.0f / (1.0f + __expf(-2.0f * x)) - 1.0f; }

__device__ __forceinline__ v8f wmma_f16(v16h a, v16h b, v8f c) {
  // (neg_a, A, neg_b, B, c_mod, C, reuse_a, reuse_b)
  return __builtin_amdgcn_wmma_f32_16x16x32_f16(false, a, false, b, (short)0, c, false, false);
}

__device__ __forceinline__ v16h pack16(v8h lo, v8h hi8) {
  v16h r;
#pragma unroll
  for (int e = 0; e < 8; ++e) { r[e] = lo[e]; r[e + 8] = hi8[e]; }
  return r;
}

// ---------------------------------------------------------------------------
// Kernel 1: GA[m, n] = sum_k x[m,k] * WA1[n,k] + bA1[n] + bA2[n]   (f16 out)
// One 16x16 output tile per wave, K = 128 in 4 chunks of 32.
// ---------------------------------------------------------------------------
__global__ __launch_bounds__(256) void ga_gemm(
    const float* __restrict__ x, const float* __restrict__ WA1,
    const float* __restrict__ bA1, const float* __restrict__ bA2,
    _Float16* __restrict__ GA)
{
  const int lane = threadIdx.x & 31;
  const int wave = threadIdx.x >> 5;
  const int ln   = lane & 15;
  const int hi   = (lane >= 16) ? 1 : 0;
  const int m0   = blockIdx.x * 16;
  const int n0   = (blockIdx.y * 8 + wave) * 16;

  const float bsum = bA1[n0 + ln] + bA2[n0 + ln];
  v8f acc;
#pragma unroll
  for (int v = 0; v < 8; ++v) acc[v] = bsum;

#pragma unroll
  for (int c = 0; c < 4; ++c) {
    // A operand: 16x32 f16 tile of x. lane -> row M=lane%16.
    // lane<16: elems 0..15 => K=32c+0..15; lane>=16: K=32c+16..31.
    v16h a;
    const int ab = 32 * c + (hi ? 16 : 0);
    const float* xr = x + (size_t)(m0 + ln) * HID;
#pragma unroll
    for (int e = 0; e < 8; ++e) {
      a[e]     = (_Float16)xr[ab + e];
      a[e + 8] = (_Float16)xr[ab + 8 + e];
    }
    // B operand: 32x16 tile of WA1^T. lane -> col N=lane%16,
    // lane<16: K=32c+0..15, lane>=16: K=32c+16..31 (contiguous in WA1 row).
    v16h b;
    const float* wr = WA1 + (size_t)(n0 + ln) * HID + 32 * c + (hi ? 16 : 0);
#pragma unroll
    for (int e = 0; e < 16; ++e) b[e] = (_Float16)wr[e];

    acc = wmma_f16(a, b, acc);
  }

#pragma unroll
  for (int v = 0; v < 8; ++v) {
    const int m = m0 + v + (hi ? 8 : 0);
    GA[(size_t)m * GATES + n0 + ln] = (_Float16)acc[v];
  }
}

// ---------------------------------------------------------------------------
// Kernel 2: sequential LSTM scan + FC. 1 block x 1024 threads (32 wave32s).
// Wave w owns gate columns [16w, 16w+16). Threads 0..127 own cell state j.
// ---------------------------------------------------------------------------
__global__ __launch_bounds__(1024) void lstm_scan(
    const _Float16* __restrict__ GA,
    const float* __restrict__ hA0, const float* __restrict__ cA0,
    const float* __restrict__ hB0, const float* __restrict__ cB0,
    const float* __restrict__ WA2, const float* __restrict__ WB1,
    const float* __restrict__ WB2,
    const float* __restrict__ bB1, const float* __restrict__ bB2,
    const float* __restrict__ Wfc, const float* __restrict__ bfc,
    float* __restrict__ out)
{
  __shared__ __align__(16) _Float16 WA2l[GATES * WPITCH];   // 136 KB
  __shared__ __align__(16) _Float16 WB1l[GATES * WPITCH];   // 136 KB
  __shared__ __align__(16) _Float16 Wfcl[VOCAB * WPITCH];   //  34 KB
  __shared__ float gates[GATES];
  __shared__ __align__(16) _Float16 hAl[HID];
  __shared__ __align__(16) _Float16 hBl[HID];

  const int tid  = threadIdx.x;
  const int lane = tid & 31;
  const int wave = tid >> 5;
  const int ln   = lane & 15;
  const int hi   = (lane >= 16) ? 1 : 0;
  const int ncol = wave * 16 + ln;

  // Stage f16 weights into LDS (padded pitch).
  for (int i = tid; i < GATES * HID; i += 1024) {
    const int n = i >> 7, k = i & 127;
    WA2l[n * WPITCH + k] = (_Float16)WA2[i];
    WB1l[n * WPITCH + k] = (_Float16)WB1[i];
  }
  for (int i = tid; i < VOCAB * HID; i += 1024) {
    const int n = i >> 7, k = i & 127;
    Wfcl[n * WPITCH + k] = (_Float16)Wfc[i];
  }

  // WB2 resident in registers as WMMA B operands (32 VGPRs/wave).
  v16h wb2[4];
#pragma unroll
  for (int c = 0; c < 4; ++c) {
    const float* wr = WB2 + (size_t)ncol * HID + 32 * c + (hi ? 16 : 0);
#pragma unroll
    for (int e = 0; e < 16; ++e) wb2[c][e] = (_Float16)wr[e];
  }

  const float bB = bB1[ncol] + bB2[ncol];
  float bfc_r = 0.0f;
  if (wave < 8) bfc_r = bfc[(wave & 7) * 16 + ln];

  // Per-thread recurrent state (threads 0..127 own hidden unit `tid`).
  float cAr = 0.f, cBr = 0.f, hAr = 0.f, hBr = 0.f;
  if (tid < HID) {
    hAr = hA0[tid]; cAr = cA0[tid]; hBr = hB0[tid]; cBr = cB0[tid];
    hAl[tid] = (_Float16)hAr;
    hBl[tid] = (_Float16)hBr;
  }
  __syncthreads();

  const v8h* hAp    = (const v8h*)hAl;
  const v8h* hBp    = (const v8h*)hBl;
  const v8h* wa2row = (const v8h*)&WA2l[ncol * WPITCH];
  const v8h* wb1row = (const v8h*)&WB1l[ncol * WPITCH];
  const v8h* wfcrow = (const v8h*)&Wfcl[((wave & 7) * 16 + ln) * WPITCH];

#pragma unroll 1
  for (int t = 0; t < SEQ; ++t) {
    // ---- Phase A: gatesA = GA[t] + hA @ WA2^T (broadcast-A trick: all 16
    //      rows of A = hA, so row 0 of D holds the matvec; C[0] seeds GA).
    {
      v8f acc = {};
      acc[0] = (float)GA[(size_t)t * GATES + ncol];
      if (lane == 0 && t + 1 < SEQ)
        __builtin_prefetch(GA + (size_t)(t + 1) * GATES + wave * 16, 0, 0);
#pragma unroll
      for (int c = 0; c < 4; ++c) {
        const v16h a = pack16(hAp[4 * c + 2 * hi], hAp[4 * c + 2 * hi + 1]);
        const v16h b = pack16(wa2row[4 * c + 2 * hi], wa2row[4 * c + 2 * hi + 1]);
        acc = wmma_f16(a, b, acc);
      }
      if (lane < 16) gates[wave * 16 + lane] = acc[0];
    }
    __syncthreads();

    // ---- Cell A activations (threads 0..127 = waves 0..3, EXEC uniform).
    if (tid < HID) {
      const float ig = gates[tid], fg = gates[HID + tid];
      const float gg = gates[2 * HID + tid], og = gates[3 * HID + tid];
      cAr = sigm(fg) * cAr + sigm(ig) * tanhfast(gg);
      hAr = sigm(og) * tanhfast(cAr);
      hAl[tid] = (_Float16)hAr;
    }
    __syncthreads();

    // ---- Phase B: gatesB = bB + hA @ WB1^T + hB @ WB2^T
    {
      v8f acc = {};
      acc[0] = bB;
#pragma unroll
      for (int c = 0; c < 4; ++c) {
        const v16h a = pack16(hAp[4 * c + 2 * hi], hAp[4 * c + 2 * hi + 1]);
        const v16h b = pack16(wb1row[4 * c + 2 * hi], wb1row[4 * c + 2 * hi + 1]);
        acc = wmma_f16(a, b, acc);
      }
#pragma unroll
      for (int c = 0; c < 4; ++c) {
        const v16h a = pack16(hBp[4 * c + 2 * hi], hBp[4 * c + 2 * hi + 1]);
        acc = wmma_f16(a, wb2[c], acc);
      }
      if (lane < 16) gates[wave * 16 + lane] = acc[0];
    }
    __syncthreads();

    // ---- Cell B activations.
    if (tid < HID) {
      const float ig = gates[tid], fg = gates[HID + tid];
      const float gg = gates[2 * HID + tid], og = gates[3 * HID + tid];
      cBr = sigm(fg) * cBr + sigm(ig) * tanhfast(gg);
      hBr = sigm(og) * tanhfast(cBr);
      hBl[tid] = (_Float16)hBr;
    }
    __syncthreads();

    // ---- FC: logits[t] = hB @ Wfc^T + bfc (waves 0..7; overlaps next PhaseA).
    if (wave < 8) {
      v8f acc = {};
      acc[0] = bfc_r;
#pragma unroll
      for (int c = 0; c < 4; ++c) {
        const v16h a = pack16(hBp[4 * c + 2 * hi], hBp[4 * c + 2 * hi + 1]);
        const v16h b = pack16(wfcrow[4 * c + 2 * hi], wfcrow[4 * c + 2 * hi + 1]);
        acc = wmma_f16(a, b, acc);
      }
      if (lane < 16) out[(size_t)t * VOCAB + wave * 16 + lane] = acc[0];
    }
  }

  // Final carry states: hA, cA, hB, cB appended after logits.
  if (tid < HID) {
    const size_t base = (size_t)SEQ * VOCAB;
    out[base + tid]           = hAr;
    out[base + HID + tid]     = cAr;
    out[base + 2 * HID + tid] = hBr;
    out[base + 3 * HID + tid] = cBr;
  }
}

// ---------------------------------------------------------------------------
extern "C" void kernel_launch(void* const* d_in, const int* in_sizes, int n_in,
                              void* d_out, int out_size, void* d_ws, size_t ws_size,
                              hipStream_t stream) {
  const float* x   = (const float*)d_in[0];
  const float* hA  = (const float*)d_in[1];
  const float* cA  = (const float*)d_in[2];
  const float* hB  = (const float*)d_in[3];
  const float* cB  = (const float*)d_in[4];
  const float* WA1 = (const float*)d_in[5];
  const float* bA1 = (const float*)d_in[6];
  const float* WA2 = (const float*)d_in[7];
  const float* bA2 = (const float*)d_in[8];
  const float* WB1 = (const float*)d_in[9];
  const float* bB1 = (const float*)d_in[10];
  const float* WB2 = (const float*)d_in[11];
  const float* bB2 = (const float*)d_in[12];
  const float* Wfc = (const float*)d_in[13];
  const float* bfc = (const float*)d_in[14];
  float* out = (float*)d_out;

  _Float16* GA = (_Float16*)d_ws;  // 65536*512*2 = 64 MB scratch

  ga_gemm<<<dim3(SEQ / 16, 4), 256, 0, stream>>>(x, WA1, bA1, bA2, GA);
  lstm_scan<<<1, 1024, 0, stream>>>(GA, hA, cA, hB, cB, WA2, WB1, WB2,
                                    bB1, bB2, Wfc, bfc, out);
}
